// ModuleNet_9672266351161
// MI455X (gfx1250) — matrix operand order repaired and checked by reading, not compiled, not run on hardware
//
#include <hip/hip_runtime.h>

typedef _Float16 f16;
typedef __attribute__((ext_vector_type(16))) _Float16 v16h;
typedef __attribute__((ext_vector_type(8)))  _Float16 v8h;
typedef __attribute__((ext_vector_type(8)))  float    v8f;

#define NB     4096
#define NP     32
#define EMBED  128
#define DIN    768   // 2*EMBED*3
#define C1     128
#define C2     32

// ---------------- workspace layout (bytes) ----------------
// [0, 6291456)               : h as f16, row-major [4096][768]
// [6291456, 6488064)         : W1 packed B-fragments (24 kt * 8 nt * 32 lanes * 16 halves)
// [6488064, 6496256)         : W2 packed B-fragments (4 kt * 2 nt * 32 lanes * 16 halves)
#define OFF_H    0
#define OFF_W1P  6291456
#define OFF_W2P  6488064

// ---------------------------------------------------------------------------
// Kernel 0: convert W1/W2 to f16, pre-swizzled into WMMA B-fragment order.
// B-matrix layout (16-bit, 32x16): lane l -> col = l&15, k = 8*(l>>4) + (e&7) + 16*(e>>3)
// within a 32-deep k-tile. We store each lane's 16 halves contiguously.
// ---------------------------------------------------------------------------
__global__ void pack_weights(const float* __restrict__ W1,
                             const float* __restrict__ W2,
                             f16* __restrict__ W1p, f16* __restrict__ W2p) {
    int idx = blockIdx.x * blockDim.x + threadIdx.x;
    const int n1 = 24 * 8 * 32 * 16;        // 98304
    if (idx < n1) {
        int e    = idx & 15;
        int lane = (idx >> 4) & 31;
        int frag = idx >> 9;
        int nt   = frag & 7;
        int kt   = frag >> 3;
        int k    = kt * 32 + 8 * (lane >> 4) + (e & 7) + 16 * (e >> 3);
        int n    = nt * 16 + (lane & 15);
        W1p[idx] = (f16)W1[k * C1 + n];
    } else {
        int t = idx - n1;
        if (t < 4 * 2 * 32 * 16) {          // 4096
            int e    = t & 15;
            int lane = (t >> 4) & 31;
            int frag = t >> 9;
            int nt   = frag & 1;
            int kt   = frag >> 1;
            int k    = kt * 32 + 8 * (lane >> 4) + (e & 7) + 16 * (e >> 3);
            int n    = nt * 16 + (lane & 15);
            W2p[t] = (f16)W2[k * C2 + n];
        }
    }
}

// ---------------------------------------------------------------------------
// Kernel 1: path gather/elementwise phase. One block per (b, group).
// 8 waves per block; wave w handles paths {w, w+8, w+16, w+24}; each lane
// owns 4 embed dims (float4). Cross-wave reduction in LDS, h written as f16.
// ---------------------------------------------------------------------------
__device__ __forceinline__ float4 ld4(const float* p) { return *(const float4*)p; }

__device__ __forceinline__ float4 step_relu(float4 x, float4 w, float4 y, float4 bm) {
    float4 r;
    r.x = fmaxf(x.x * w.x * y.x + bm.x, 0.f);
    r.y = fmaxf(x.y * w.y * y.y + bm.y, 0.f);
    r.z = fmaxf(x.z * w.z * y.z + bm.z, 0.f);
    r.w = fmaxf(x.w * w.w * y.w + bm.w, 0.f);
    return r;
}

__global__ void paths_kernel(const int* __restrict__ ent1, const int* __restrict__ mid1,
                             const float* __restrict__ cnt1,
                             const int* __restrict__ ent2, const int* __restrict__ mid2,
                             const float* __restrict__ cnt2,
                             const int* __restrict__ ent3, const int* __restrict__ mid3,
                             const float* __restrict__ cnt3,
                             const float* __restrict__ E, const float* __restrict__ W,
                             const float* __restrict__ Bm,
                             f16* __restrict__ hf16) {
    const int b = blockIdx.x;
    const int g = blockIdx.y;          // group 0..2, L = g+1
    const int L = g + 1;

    const int*   ent; const int* mid; const float* cnt;
    if (g == 0)      { ent = ent1; mid = mid1; cnt = cnt1; }
    else if (g == 1) { ent = ent2; mid = mid2; cnt = cnt2; }
    else             { ent = ent3; mid = mid3; cnt = cnt3; }

    const int tid  = threadIdx.x;
    const int wave = tid >> 5;
    const int lane = tid & 31;
    const int d4   = lane * 4;         // 4 embed dims per lane

    __shared__ float red[8 * 256];
    __shared__ float csw[8];

    float4 acc1 = {0.f, 0.f, 0.f, 0.f};
    float4 acc2 = {0.f, 0.f, 0.f, 0.f};
    float  csum = 0.f;

    #pragma unroll
    for (int pi = 0; pi < 4; ++pi) {
        const int p  = pi * 8 + wave;
        const int bp = b * NP + p;
        const float c = cnt[bp];

        const int* ep = ent + (size_t)bp * (L + 1);
        const int* mp = mid + (size_t)bp * L;

        float4 x = ld4(E + (size_t)ep[0] * EMBED + d4);
        float4 y = x;
        int ms[3];
        for (int i = 0; i < L; ++i) {
            ms[i] = mp[i];
            y = ld4(E + (size_t)ep[i + 1] * EMBED + d4);
            float4 w  = ld4(W  + (size_t)ms[i] * EMBED + d4);
            float4 bm = ld4(Bm + (size_t)ms[i] * EMBED + d4);
            x = step_relu(x, w, y, bm);
        }
        // forward output
        acc1.x += c * x.x; acc1.y += c * x.y; acc1.z += c * x.z; acc1.w += c * x.w;

        // backward direction with stale y (== E[ents[L]]), x2 starts at y
        float4 x2 = y;
        for (int i = L - 1; i >= 0; --i) {
            float4 w  = ld4(W  + (size_t)ms[i] * EMBED + d4);
            float4 bm = ld4(Bm + (size_t)ms[i] * EMBED + d4);
            x2 = step_relu(x2, w, y, bm);
        }
        acc2.x += c * x2.x; acc2.y += c * x2.y; acc2.z += c * x2.z; acc2.w += c * x2.w;
        csum += c;
    }

    float* r = red + wave * 256;
    r[d4 + 0]       = acc1.x; r[d4 + 1]       = acc1.y;
    r[d4 + 2]       = acc1.z; r[d4 + 3]       = acc1.w;
    r[128 + d4 + 0] = acc2.x; r[128 + d4 + 1] = acc2.y;
    r[128 + d4 + 2] = acc2.z; r[128 + d4 + 3] = acc2.w;
    if (lane == 0) csw[wave] = csum;
    __syncthreads();

    // tid in [0,256): reduce dim tid across the 8 waves
    float num = 0.f;
    #pragma unroll
    for (int w = 0; w < 8; ++w) num += red[w * 256 + tid];
    float den = 0.f;
    #pragma unroll
    for (int w = 0; w < 8; ++w) den += csw[w];

    hf16[(size_t)b * DIN + g * 256 + tid] = (f16)(num / den);
}

// ---------------------------------------------------------------------------
// Kernel 2: fused MLP via v_wmma_f32_16x16x32_f16. One wave per 16-row tile.
// A layout (16-bit 16x32): lane l -> row = l&15; elems 0..7 at k = 8*(l>>4),
// elems 8..15 at k = 8*(l>>4)+16  (two contiguous 16B runs).
// C/D layout (f32 16x16): lane l -> col = l&15, reg r -> row = r + 8*(l>>4).
// ---------------------------------------------------------------------------
__device__ __forceinline__ v16h load_a16(const f16* p) {
    v8h lo = *(const v8h*)(p);
    v8h hi = *(const v8h*)(p + 16);
    v16h a;
    #pragma unroll
    for (int j = 0; j < 8; ++j) { a[j] = lo[j]; a[j + 8] = hi[j]; }
    return a;
}

__global__ void mlp_kernel(const f16* __restrict__ hf16,
                           const f16* __restrict__ W1p, const f16* __restrict__ W2p,
                           const float* __restrict__ b1, const float* __restrict__ b2,
                           const float* __restrict__ W3, const float* __restrict__ b3,
                           float* __restrict__ out) {
    const int lane = threadIdx.x;        // 0..31, single wave per block
    const int m0   = blockIdx.x * 16;
    const int row  = lane & 15;
    const int half = lane >> 4;
    const int col  = lane & 15;

    __shared__ __align__(32) f16  act1[16 * C1];   // layer-1 relu output, f16
    __shared__ __align__(16) float act2[16 * C2];  // layer-2 relu output, f32

    // ---- layer 1: [16 x 768] @ [768 x 128] ----
    v8f acc[8] = {};
    for (int kt = 0; kt < 24; ++kt) {
        const f16* ap = hf16 + (size_t)(m0 + row) * DIN + kt * 32 + half * 8;
        v16h a = load_a16(ap);
        #pragma unroll
        for (int nt = 0; nt < 8; ++nt) {
            v16h bf = *(const v16h*)(W1p + ((size_t)(kt * 8 + nt) * 32 + lane) * 16);
            acc[nt] = __builtin_amdgcn_wmma_f32_16x16x32_f16(
                false, a, false, bf, (short)0, acc[nt], false, false);
        }
    }
    #pragma unroll
    for (int nt = 0; nt < 8; ++nt) {
        float bias = b1[nt * 16 + col];
        #pragma unroll
        for (int r = 0; r < 8; ++r) {
            float v = fmaxf(acc[nt][r] + bias, 0.f);
            act1[(r + 8 * half) * C1 + nt * 16 + col] = (f16)v;
        }
    }
    __syncthreads();   // single-wave WG: lowers to s_nop; DS same-wave order suffices

    // ---- layer 2: [16 x 128] @ [128 x 32] ----
    v8f acc2[2] = {};
    #pragma unroll
    for (int kt = 0; kt < 4; ++kt) {
        const f16* ap = act1 + row * C1 + kt * 32 + half * 8;
        v16h a = load_a16(ap);
        #pragma unroll
        for (int nt = 0; nt < 2; ++nt) {
            v16h bf = *(const v16h*)(W2p + ((size_t)(kt * 2 + nt) * 32 + lane) * 16);
            acc2[nt] = __builtin_amdgcn_wmma_f32_16x16x32_f16(
                false, a, false, bf, (short)0, acc2[nt], false, false);
        }
    }
    #pragma unroll
    for (int nt = 0; nt < 2; ++nt) {
        float bias = b2[nt * 16 + col];
        #pragma unroll
        for (int r = 0; r < 8; ++r) {
            float v = fmaxf(acc2[nt][r] + bias, 0.f);
            act2[(r + 8 * half) * C2 + nt * 16 + col] = v;
        }
    }
    __syncthreads();

    // ---- layer 3: [16 x 32] @ [32 x 1] + b3 ----
    if (lane < 16) {
        float s = b3[0];
        #pragma unroll
        for (int c = 0; c < C2; ++c) s += act2[lane * C2 + c] * W3[c];
        out[m0 + lane] = s;
    }
}

// ---------------------------------------------------------------------------
extern "C" void kernel_launch(void* const* d_in, const int* in_sizes, int n_in,
                              void* d_out, int out_size, void* d_ws, size_t ws_size,
                              hipStream_t stream) {
    const int*   ent1 = (const int*)d_in[0];
    const int*   mid1 = (const int*)d_in[1];
    const float* cnt1 = (const float*)d_in[2];
    const int*   ent2 = (const int*)d_in[3];
    const int*   mid2 = (const int*)d_in[4];
    const float* cnt2 = (const float*)d_in[5];
    const int*   ent3 = (const int*)d_in[6];
    const int*   mid3 = (const int*)d_in[7];
    const float* cnt3 = (const float*)d_in[8];
    const float* E    = (const float*)d_in[9];
    const float* W    = (const float*)d_in[10];
    const float* Bm   = (const float*)d_in[11];
    const float* W1   = (const float*)d_in[12];
    const float* b1   = (const float*)d_in[13];
    const float* W2   = (const float*)d_in[14];
    const float* b2   = (const float*)d_in[15];
    const float* W3   = (const float*)d_in[16];
    const float* b3   = (const float*)d_in[17];

    char* ws   = (char*)d_ws;
    f16*  hf16 = (f16*)(ws + OFF_H);
    f16*  W1p  = (f16*)(ws + OFF_W1P);
    f16*  W2p  = (f16*)(ws + OFF_W2P);

    // 0) pack weights to f16 B-fragments
    const int npack = 24 * 8 * 32 * 16 + 4 * 2 * 32 * 16;
    pack_weights<<<(npack + 255) / 256, 256, 0, stream>>>(W1, W2, W1p, W2p);

    // 1) gather/path phase -> h[4096,768] f16
    paths_kernel<<<dim3(NB, 3), 256, 0, stream>>>(ent1, mid1, cnt1,
                                                  ent2, mid2, cnt2,
                                                  ent3, mid3, cnt3,
                                                  E, W, Bm, hf16);

    // 2) fused WMMA MLP -> out[4096]
    mlp_kernel<<<NB / 16, 32, 0, stream>>>(hf16, W1p, W2p, b1, b2, W3, b3,
                                           (float*)d_out);
}
